// CVNet_19842748907580
// MI455X (gfx1250) — matrix operand order repaired and checked by reading, not compiled, not run on hardware
//
#include <hip/hip_runtime.h>
#include <hip/hip_bf16.h>

typedef _Float16 v16h __attribute__((ext_vector_type(16)));
typedef _Float16 v8h  __attribute__((ext_vector_type(8)));
typedef _Float16 v4h  __attribute__((ext_vector_type(4)));
typedef float    v8f  __attribute__((ext_vector_type(8)));

#define B_       16384
#define IN_      64
#define HID_     128
#define E_       64
#define ROWS_    32               // samples per wave (2 M-tiles)
#define MAX_TILES 576             // 16384/32 + 64 (per-expert padding to 32)
#define MAX_SLOTS (MAX_TILES * ROWS_)   // 18432

// ---- workspace layout (int units) ----
// [0,64)      counts
// [64,128)    cursors
// [128,193)   offsets (65)
// [193]       nTiles
// [256,832)   tileExpert
// [2048,2048+MAX_SLOTS)   perm (sample index or -1)   -> ends at 20480
// [20480,20480+B_)        eArr
// [40960, +E_*HID_*IN_/2) w1h: f16 W1 transposed per expert [E][128][64]
#define WS_COUNT   0
#define WS_CURSOR  64
#define WS_OFF     128
#define WS_NTILES  193
#define WS_TILEEXP 256
#define WS_PERM    2048
#define WS_EARR    20480
#define WS_W1H     40960          // 32B-aligned (int offset multiple of 8)

__global__ void k_init(int* ws) {
    int i = blockIdx.x * blockDim.x + threadIdx.x;
    if (i < 128) ws[i] = 0;                       // counts + cursors
    if (i < MAX_SLOTS) ws[WS_PERM + i] = -1;      // perm padding sentinel
}

__global__ void k_expert(const long long* __restrict__ num,
                         const long long* __restrict__ c,
                         int* __restrict__ ws) {
    int b = blockIdx.x * blockDim.x + threadIdx.x;
    if (b >= B_) return;
    int e = (int)c[num[b]];
    ws[WS_EARR + b] = e;
    atomicAdd(&ws[WS_COUNT + e], 1);
}

__global__ void k_offsets(int* ws) {
    if (blockIdx.x != 0 || threadIdx.x != 0) return;
    int run = 0;
    for (int e = 0; e < E_; ++e) {
        ws[WS_OFF + e] = run;
        int pc = (ws[WS_COUNT + e] + (ROWS_ - 1)) & ~(ROWS_ - 1);  // pad to 32
        int tb = run / ROWS_;
        for (int t = 0; t < pc / ROWS_; ++t) ws[WS_TILEEXP + tb + t] = e;
        run += pc;
    }
    ws[WS_OFF + E_] = run;
    ws[WS_NTILES]   = run / ROWS_;
}

__global__ void k_scatter(int* __restrict__ ws) {
    int b = blockIdx.x * blockDim.x + threadIdx.x;
    if (b >= B_) return;
    int e   = ws[WS_EARR + b];
    int pos = ws[WS_OFF + e] + atomicAdd(&ws[WS_CURSOR + e], 1);
    ws[WS_PERM + pos] = b;
}

// W1 [E][64][128] f32 (k-major)  ->  w1h [E][128][64] f16 (n-major)
__global__ void k_convert(const float* __restrict__ W1, _Float16* __restrict__ w1h) {
    int idx = blockIdx.x * blockDim.x + threadIdx.x;   // E*128*64 = 524288
    if (idx >= E_ * HID_ * IN_) return;
    int e = idx >> 13;        // / 8192
    int r = idx & 8191;
    int n = r >> 6;           // / 64
    int k = r & 63;
    w1h[idx] = (_Float16)W1[(size_t)e * IN_ * HID_ + (size_t)k * HID_ + n];
}

__global__ void __launch_bounds__(32)
k_moe(const float* __restrict__ x,  const _Float16* __restrict__ w1h,
      const float* __restrict__ b1, const float* __restrict__ W2,
      const float* __restrict__ b2, const int* __restrict__ ws,
      float* __restrict__ out) {
    __shared__ _Float16 xs[ROWS_ * IN_];   // 4 KB, [32 rows][64 k] f16

    const int t = blockIdx.x;
    if (t >= ws[WS_NTILES]) return;        // wave-uniform: EXEC stays all-1s

    const int lane = threadIdx.x;
    const int m    = lane & 15;
    const int hi   = lane >> 4;
    const int e    = ws[WS_TILEEXP + t];
    const int* perm = ws + WS_PERM + t * ROWS_;
    const _Float16* w1e = w1h + (size_t)e * HID_ * IN_;   // [128][64] n-major

    // ---- stage 32 x-rows -> LDS f16 (zeros for padded slots) ----
    for (int i = lane; i < ROWS_ * IN_ / 4; i += 32) {
        int r  = i >> 4;               // 16 float4 per 64-elem row
        int c4 = i & 15;
        int idx = perm[r];
        float4 v = make_float4(0.f, 0.f, 0.f, 0.f);
        if (idx >= 0) v = ((const float4*)(x + (size_t)idx * IN_))[c4];
        v4h h = { (_Float16)v.x, (_Float16)v.y, (_Float16)v.z, (_Float16)v.w };
        *(v4h*)(xs + r * IN_ + 4 * c4) = h;    // 8B-aligned ds_store_b64
    }
    // ---- per-lane column constants for layer 2 (col = nt*16 + m) ----
    float b1v[8], w2v[8];
    #pragma unroll
    for (int nt = 0; nt < 8; ++nt) {
        int col = nt * 16 + m;
        b1v[nt] = b1[(size_t)e * HID_ + col];
        w2v[nt] = W2[(size_t)e * HID_ + col];
    }
    __syncthreads();

    // ---- layer 1: two 16x64 A-tiles x one 64x128 B, B reused across tiles ----
    v8f acc[2][8] = {};
    #pragma unroll
    for (int kc = 0; kc < 2; ++kc) {
        // A operands (16-bit A 16x32): per lane 16 contiguous
        // halves: k = kc*32 + hi*16 + [0,16) (matches B)
        v16h a[2];
        #pragma unroll
        for (int mt = 0; mt < 2; ++mt) {
            const v8h* xr = (const v8h*)(xs + (mt * 16 + m) * IN_ + kc * 32);
            v8h alo = xr[2 * hi];      // 16B-aligned ds_load_b128
            v8h ahi = xr[2 * hi + 1];
            a[mt] = __builtin_shufflevector(alo, ahi,
                    0,1,2,3,4,5,6,7,8,9,10,11,12,13,14,15);
        }
        #pragma unroll
        for (int nt = 0; nt < 8; ++nt) {
            // B operand: col n = nt*16+m, 16 contiguous k at kc*32 + hi*16
            int n = nt * 16 + m;
            v16h bb = *(const v16h*)(w1e + (size_t)n * IN_ + kc * 32 + hi * 16);
            acc[0][nt] = __builtin_amdgcn_wmma_f32_16x16x32_f16(
                false, a[0], false, bb, (short)0, acc[0][nt], false, false);
            acc[1][nt] = __builtin_amdgcn_wmma_f32_16x16x32_f16(
                false, a[1], false, bb, (short)0, acc[1][nt], false, false);
        }
    }

    // ---- layer 2 fused in-register: lane holds h[mt*16+j+8*hi][nt*16+m] ----
    #pragma unroll
    for (int mt = 0; mt < 2; ++mt) {
        #pragma unroll
        for (int j = 0; j < 8; ++j) {
            float p = 0.f;
            #pragma unroll
            for (int nt = 0; nt < 8; ++nt) {
                float h = acc[mt][nt][j] + b1v[nt];
                p += (h > 0.f ? h : 0.f) * w2v[nt];
            }
            // butterfly-reduce over the 16 m-lanes (stays within each half-wave)
            p += __shfl_xor(p, 1);
            p += __shfl_xor(p, 2);
            p += __shfl_xor(p, 4);
            p += __shfl_xor(p, 8);
            if (m == 0) {
                int row = j + hi * 8;
                int idx = perm[mt * 16 + row];
                if (idx >= 0) {
                    float y = b2[e] + p;
                    out[idx] = 1.f / (1.f + __expf(-y));
                }
            }
        }
    }
}

extern "C" void kernel_launch(void* const* d_in, const int* in_sizes, int n_in,
                              void* d_out, int out_size, void* d_ws, size_t ws_size,
                              hipStream_t stream) {
    const float*     x   = (const float*)d_in[0];
    const long long* num = (const long long*)d_in[1];
    const long long* c   = (const long long*)d_in[2];
    const float*     W1  = (const float*)d_in[3];
    const float*     b1  = (const float*)d_in[4];
    const float*     W2  = (const float*)d_in[5];
    const float*     b2  = (const float*)d_in[6];
    float* out = (float*)d_out;
    int*   ws  = (int*)d_ws;
    _Float16* w1h = (_Float16*)(ws + WS_W1H);

    k_init   <<<(MAX_SLOTS + 255) / 256,        256, 0, stream>>>(ws);
    k_expert <<<(B_ + 255) / 256,               256, 0, stream>>>(num, c, ws);
    k_offsets<<<1,                               32, 0, stream>>>(ws);
    k_scatter<<<(B_ + 255) / 256,               256, 0, stream>>>(ws);
    k_convert<<<(E_ * HID_ * IN_ + 255) / 256,  256, 0, stream>>>(W1, w1h);
    k_moe    <<<MAX_TILES,                       32, 0, stream>>>(x, w1h, b1, W2, b2, ws, out);
}
